// AttentionCritic_46849503265097
// MI455X (gfx1250) — matrix operand order repaired and checked by reading, not compiled, not run on hardware
//
#include <hip/hip_runtime.h>

typedef __attribute__((ext_vector_type(16))) _Float16 v16h;
typedef __attribute__((ext_vector_type(8)))  _Float16 v8h;
typedef __attribute__((ext_vector_type(8)))  float    v8f;
typedef __attribute__((ext_vector_type(4)))  _Float16 v4h;

#define N_AG   8
#define B_SZ   8192
#define S_DIM  64
#define A_DIM  16
#define H_DIM  512
#define E_HD   4
#define D_DIM  128
#define K_SA   96   // 64+16 padded to multiple of 32

__device__ __forceinline__ float lrelu_f(float x) { return x > 0.f ? x : 0.01f * x; }

// ---------------- staging kernels (all write coalesced f16) ----------------

__global__ void cvt_f16_kernel(const float* __restrict__ in, _Float16* __restrict__ out, size_t n) {
  size_t i = (size_t)blockIdx.x * blockDim.x + threadIdx.x;
  size_t stride = (size_t)gridDim.x * blockDim.x;
  for (; i < n; i += stride) out[i] = (_Float16)in[i];
}

// sa[n][b][0:64]=states, [64:80]=actions, [80:96]=0
__global__ void build_sa_kernel(const float* __restrict__ st, const float* __restrict__ ac,
                                _Float16* __restrict__ out) {
  const size_t total = (size_t)N_AG * B_SZ * K_SA;
  size_t i = (size_t)blockIdx.x * blockDim.x + threadIdx.x;
  size_t stride = (size_t)gridDim.x * blockDim.x;
  for (; i < total; i += stride) {
    size_t nb = i / K_SA;
    int c = (int)(i % K_SA);
    float x;
    if (c < S_DIM)              x = st[nb * S_DIM + c];
    else if (c < S_DIM + A_DIM) x = ac[nb * A_DIM + (c - S_DIM)];
    else                        x = 0.f;
    out[i] = (_Float16)x;
  }
}

// Ws [N,S,H] -> WsT [N,H,S]
__global__ void t_ws_kernel(const float* __restrict__ in, _Float16* __restrict__ out) {
  const size_t total = (size_t)N_AG * H_DIM * S_DIM;
  size_t i = (size_t)blockIdx.x * blockDim.x + threadIdx.x;
  size_t stride = (size_t)gridDim.x * blockDim.x;
  for (; i < total; i += stride) {
    size_t n = i / ((size_t)H_DIM * S_DIM);
    size_t rem = i % ((size_t)H_DIM * S_DIM);
    size_t h = rem / S_DIM, s = rem % S_DIM;
    out[i] = (_Float16)in[(n * S_DIM + s) * H_DIM + h];
  }
}

// Wsa [N,80,H] -> WsaT [N,H,96] zero-padded
__global__ void t_wsa_kernel(const float* __restrict__ in, _Float16* __restrict__ out) {
  const size_t total = (size_t)N_AG * H_DIM * K_SA;
  size_t i = (size_t)blockIdx.x * blockDim.x + threadIdx.x;
  size_t stride = (size_t)gridDim.x * blockDim.x;
  for (; i < total; i += stride) {
    size_t n = i / ((size_t)H_DIM * K_SA);
    size_t rem = i % ((size_t)H_DIM * K_SA);
    size_t h = rem / K_SA;
    int r = (int)(rem % K_SA);
    out[i] = (r < S_DIM + A_DIM) ? (_Float16)in[(n * 80 + r) * H_DIM + h] : (_Float16)0.f;
  }
}

// W[E,H,D] -> WT[E*D, H], row = e*128+d : WT[row][h] = W[e][h][d]
__global__ void t_headw_kernel(const float* __restrict__ in, _Float16* __restrict__ out) {
  const size_t total = (size_t)H_DIM * H_DIM;
  size_t i = (size_t)blockIdx.x * blockDim.x + threadIdx.x;
  size_t stride = (size_t)gridDim.x * blockDim.x;
  for (; i < total; i += stride) {
    size_t row = i / H_DIM, h = i % H_DIM;
    size_t e = row >> 7, d = row & 127;
    out[i] = (_Float16)in[(e * H_DIM + h) * D_DIM + d];
  }
}

// W [N,2H,H] -> WT [N,H,2H]
__global__ void t_w2h_kernel(const float* __restrict__ in, _Float16* __restrict__ out) {
  const size_t total = (size_t)N_AG * H_DIM * 2 * H_DIM;
  size_t i = (size_t)blockIdx.x * blockDim.x + threadIdx.x;
  size_t stride = (size_t)gridDim.x * blockDim.x;
  for (; i < total; i += stride) {
    size_t n = i / ((size_t)H_DIM * 2 * H_DIM);
    size_t rem = i % ((size_t)H_DIM * 2 * H_DIM);
    size_t c = rem / (2 * H_DIM), k = rem % (2 * H_DIM);
    out[i] = (_Float16)in[(n * 2 * H_DIM + k) * H_DIM + c];
  }
}

// ---------------- WMMA GEMM: C = act(A @ W + bias) ----------------------------
// WT = W transposed: [Ncols][K] row-major -> B fragments are two contiguous 16B loads per lane.
// A split in two K-segments (concat inputs): A1 covers [0,K1), A2 covers [K1,K).
// Each wave computes a 64x32 tile: 4 A-frags x 2 B-frags -> 8 wmma per k-step,
// with pointer-increment addressing (no per-iteration 64-bit address rebuilds).
__global__ __launch_bounds__(256)
void gemm_wmma_f16(const _Float16* __restrict__ A1, size_t sA1,
                   const _Float16* __restrict__ A2, size_t sA2,
                   int K1, int K,
                   const _Float16* __restrict__ WT, size_t sW,
                   const float* __restrict__ bias, size_t sBias,
                   _Float16* __restrict__ C, size_t sC,
                   int Brows, int Ncols, int act)
{
  const int ag = blockIdx.y;
  A1 += (size_t)ag * sA1;
  if (A2)   A2   += (size_t)ag * sA2;
  WT += (size_t)ag * sW;
  if (bias) bias += (size_t)ag * sBias;
  C  += (size_t)ag * sC;

  const int lane = threadIdx.x;                 // wave32
  const int tilesN = Ncols >> 5;
  const int t = blockIdx.x * 8 + threadIdx.y;
  const int tm = t / tilesN, tn = t % tilesN;
  const int m0 = tm * 64, n0 = tn * 32;
  if (m0 >= Brows) return;
  const int hv  = lane >> 4;
  const int col = lane & 15;
  const int K2 = K - K1;

  // B fragment streams: lane reads WT[n0(+16)+col][k0 + 8h .. +7] and [k0 + 16 + 8h .. +7], two 16B loads
  const _Float16* q0 = WT + (size_t)(n0 + col) * K + 8 * hv;
  const _Float16* q1 = WT + (size_t)(n0 + 16 + col) * K + 8 * hv;

  v8f acc[8];
  #pragma unroll
  for (int i = 0; i < 8; ++i) acc[i] = (v8f){};

  // ---- segment 1: A1, k in [0, K1) ----
  {
    const _Float16* p0 = A1 + (size_t)(m0 +  0 + col) * K1 + 8 * hv;
    const _Float16* p1 = A1 + (size_t)(m0 + 16 + col) * K1 + 8 * hv;
    const _Float16* p2 = A1 + (size_t)(m0 + 32 + col) * K1 + 8 * hv;
    const _Float16* p3 = A1 + (size_t)(m0 + 48 + col) * K1 + 8 * hv;
    for (int k0 = 0; k0 < K1; k0 += 32) {
      __builtin_prefetch(q0 + 64, 0, 1);        // global_prefetch_b8, 2 k-steps ahead
      __builtin_prefetch(q1 + 64, 0, 1);
      v16h a0 = __builtin_shufflevector(*(const v8h*)p0, *(const v8h*)(p0 + 16),
                                        0,1,2,3,4,5,6,7,8,9,10,11,12,13,14,15);
      v16h a1 = __builtin_shufflevector(*(const v8h*)p1, *(const v8h*)(p1 + 16),
                                        0,1,2,3,4,5,6,7,8,9,10,11,12,13,14,15);
      v16h a2 = __builtin_shufflevector(*(const v8h*)p2, *(const v8h*)(p2 + 16),
                                        0,1,2,3,4,5,6,7,8,9,10,11,12,13,14,15);
      v16h a3 = __builtin_shufflevector(*(const v8h*)p3, *(const v8h*)(p3 + 16),
                                        0,1,2,3,4,5,6,7,8,9,10,11,12,13,14,15);
      v16h b0 = __builtin_shufflevector(*(const v8h*)q0, *(const v8h*)(q0 + 16),
                                        0,1,2,3,4,5,6,7,8,9,10,11,12,13,14,15);
      v16h b1 = __builtin_shufflevector(*(const v8h*)q1, *(const v8h*)(q1 + 16),
                                        0,1,2,3,4,5,6,7,8,9,10,11,12,13,14,15);
      acc[0] = __builtin_amdgcn_wmma_f32_16x16x32_f16(false, a0, false, b0, (short)0, acc[0], false, false);
      acc[1] = __builtin_amdgcn_wmma_f32_16x16x32_f16(false, a0, false, b1, (short)0, acc[1], false, false);
      acc[2] = __builtin_amdgcn_wmma_f32_16x16x32_f16(false, a1, false, b0, (short)0, acc[2], false, false);
      acc[3] = __builtin_amdgcn_wmma_f32_16x16x32_f16(false, a1, false, b1, (short)0, acc[3], false, false);
      acc[4] = __builtin_amdgcn_wmma_f32_16x16x32_f16(false, a2, false, b0, (short)0, acc[4], false, false);
      acc[5] = __builtin_amdgcn_wmma_f32_16x16x32_f16(false, a2, false, b1, (short)0, acc[5], false, false);
      acc[6] = __builtin_amdgcn_wmma_f32_16x16x32_f16(false, a3, false, b0, (short)0, acc[6], false, false);
      acc[7] = __builtin_amdgcn_wmma_f32_16x16x32_f16(false, a3, false, b1, (short)0, acc[7], false, false);
      p0 += 32; p1 += 32; p2 += 32; p3 += 32; q0 += 32; q1 += 32;
    }
  }
  // ---- segment 2: A2, k in [K1, K) ----
  if (K2 > 0) {
    const _Float16* p0 = A2 + (size_t)(m0 +  0 + col) * K2 + 8 * hv;
    const _Float16* p1 = A2 + (size_t)(m0 + 16 + col) * K2 + 8 * hv;
    const _Float16* p2 = A2 + (size_t)(m0 + 32 + col) * K2 + 8 * hv;
    const _Float16* p3 = A2 + (size_t)(m0 + 48 + col) * K2 + 8 * hv;
    for (int k0 = 0; k0 < K2; k0 += 32) {
      __builtin_prefetch(q0 + 64, 0, 1);
      __builtin_prefetch(q1 + 64, 0, 1);
      v16h a0 = __builtin_shufflevector(*(const v8h*)p0, *(const v8h*)(p0 + 16),
                                        0,1,2,3,4,5,6,7,8,9,10,11,12,13,14,15);
      v16h a1 = __builtin_shufflevector(*(const v8h*)p1, *(const v8h*)(p1 + 16),
                                        0,1,2,3,4,5,6,7,8,9,10,11,12,13,14,15);
      v16h a2 = __builtin_shufflevector(*(const v8h*)p2, *(const v8h*)(p2 + 16),
                                        0,1,2,3,4,5,6,7,8,9,10,11,12,13,14,15);
      v16h a3 = __builtin_shufflevector(*(const v8h*)p3, *(const v8h*)(p3 + 16),
                                        0,1,2,3,4,5,6,7,8,9,10,11,12,13,14,15);
      v16h b0 = __builtin_shufflevector(*(const v8h*)q0, *(const v8h*)(q0 + 16),
                                        0,1,2,3,4,5,6,7,8,9,10,11,12,13,14,15);
      v16h b1 = __builtin_shufflevector(*(const v8h*)q1, *(const v8h*)(q1 + 16),
                                        0,1,2,3,4,5,6,7,8,9,10,11,12,13,14,15);
      acc[0] = __builtin_amdgcn_wmma_f32_16x16x32_f16(false, a0, false, b0, (short)0, acc[0], false, false);
      acc[1] = __builtin_amdgcn_wmma_f32_16x16x32_f16(false, a0, false, b1, (short)0, acc[1], false, false);
      acc[2] = __builtin_amdgcn_wmma_f32_16x16x32_f16(false, a1, false, b0, (short)0, acc[2], false, false);
      acc[3] = __builtin_amdgcn_wmma_f32_16x16x32_f16(false, a1, false, b1, (short)0, acc[3], false, false);
      acc[4] = __builtin_amdgcn_wmma_f32_16x16x32_f16(false, a2, false, b0, (short)0, acc[4], false, false);
      acc[5] = __builtin_amdgcn_wmma_f32_16x16x32_f16(false, a2, false, b1, (short)0, acc[5], false, false);
      acc[6] = __builtin_amdgcn_wmma_f32_16x16x32_f16(false, a3, false, b0, (short)0, acc[6], false, false);
      acc[7] = __builtin_amdgcn_wmma_f32_16x16x32_f16(false, a3, false, b1, (short)0, acc[7], false, false);
      p0 += 32; p1 += 32; p2 += 32; p3 += 32; q0 += 32; q1 += 32;
    }
  }

  const float bb0 = bias ? bias[n0 + col]      : 0.f;
  const float bb1 = bias ? bias[n0 + 16 + col] : 0.f;
  #pragma unroll
  for (int ai = 0; ai < 4; ++ai) {
    #pragma unroll
    for (int r = 0; r < 8; ++r) {
      const int m = m0 + 16 * ai + r + 8 * hv;  // C layout: VGPR r -> M = r + 8*(lane/16)
      float x0 = acc[ai * 2 + 0][r] + bb0;
      float x1 = acc[ai * 2 + 1][r] + bb1;
      if (act) { x0 = lrelu_f(x0); x1 = lrelu_f(x1); }
      C[(size_t)m * Ncols + n0 + col]      = (_Float16)x0;
      C[(size_t)m * Ncols + n0 + 16 + col] = (_Float16)x1;
    }
  }
}

// ---------------- attention: one wave per (b,e); N=8 agents, D=128 -------------
__global__ __launch_bounds__(256)
void attention_kernel(const _Float16* __restrict__ keys,
                      const _Float16* __restrict__ sels,
                      const _Float16* __restrict__ vals,
                      _Float16* __restrict__ other)
{
  const int id = blockIdx.x * 8 + threadIdx.y;  // (b,e) pair
  const int lane = threadIdx.x;
  const int b = id >> 2, e = id & 3;
  if (b >= B_SZ) return;
  const int d0 = lane * 4;

  float key[N_AG][4], sel[N_AG][4], val[N_AG][4];
  #pragma unroll
  for (int j = 0; j < N_AG; ++j) {
    const size_t off = ((size_t)j * B_SZ + b) * H_DIM + e * D_DIM + d0;
    v4h kk = *(const v4h*)(keys + off);
    v4h ss = *(const v4h*)(sels + off);
    v4h vv = *(const v4h*)(vals + off);
    #pragma unroll
    for (int p = 0; p < 4; ++p) {
      key[j][p] = (float)kk[p]; sel[j][p] = (float)ss[p]; val[j][p] = (float)vv[p];
    }
  }

  const float scale = 0.08838834764831845f;     // 1/sqrt(128)
  #pragma unroll
  for (int i = 0; i < N_AG; ++i) {
    float logit[N_AG];
    #pragma unroll
    for (int j = 0; j < N_AG; ++j) {
      float p = sel[i][0] * key[j][0] + sel[i][1] * key[j][1]
              + sel[i][2] * key[j][2] + sel[i][3] * key[j][3];
      p += __shfl_xor(p, 16, 32);
      p += __shfl_xor(p,  8, 32);
      p += __shfl_xor(p,  4, 32);
      p += __shfl_xor(p,  2, 32);
      p += __shfl_xor(p,  1, 32);
      logit[j] = p * scale;
    }
    float mx = logit[0];
    #pragma unroll
    for (int j = 1; j < N_AG; ++j) mx = fmaxf(mx, logit[j]);
    float w[N_AG], s = 0.f;
    #pragma unroll
    for (int j = 0; j < N_AG; ++j) { w[j] = __expf(logit[j] - mx); s += w[j]; }
    const float inv = 1.f / s;
    float o[4] = {0.f, 0.f, 0.f, 0.f};
    #pragma unroll
    for (int j = 0; j < N_AG; ++j) {
      const float ww = w[j] * inv;
      #pragma unroll
      for (int p = 0; p < 4; ++p) o[p] += ww * val[j][p];
    }
    v4h out;
    #pragma unroll
    for (int p = 0; p < 4; ++p) out[p] = (_Float16)o[p];
    *(v4h*)(other + ((size_t)i * B_SZ + b) * H_DIM + e * D_DIM + d0) = out;
  }
}

// ---------------- final heads: wave per (n,b), q and v dots --------------------
__global__ __launch_bounds__(256)
void head_kernel(const _Float16* __restrict__ hidc, const _Float16* __restrict__ hidv,
                 const float* __restrict__ Wc2, const float* __restrict__ bc2,
                 const float* __restrict__ Wv2, const float* __restrict__ bv2,
                 float* __restrict__ outq, float* __restrict__ outv)
{
  const int id = blockIdx.x * 8 + threadIdx.y;  // n*B + b
  const int lane = threadIdx.x;
  if (id >= N_AG * B_SZ) return;
  const int n = id / B_SZ;
  const size_t row = (size_t)id * H_DIM;
  float aq = 0.f, av = 0.f;
  #pragma unroll 4
  for (int h = lane; h < H_DIM; h += 32) {
    aq += (float)hidc[row + h] * Wc2[n * H_DIM + h];
    av += (float)hidv[row + h] * Wv2[n * H_DIM + h];
  }
  #pragma unroll
  for (int off = 16; off; off >>= 1) {
    aq += __shfl_xor(aq, off, 32);
    av += __shfl_xor(av, off, 32);
  }
  if (lane == 0) { outq[id] = aq + bc2[n]; outv[id] = av + bv2[n]; }
}

// ---------------- host-side launch -------------------------------------------

extern "C" void kernel_launch(void* const* d_in, const int* in_sizes, int n_in,
                              void* d_out, int out_size, void* d_ws, size_t ws_size,
                              hipStream_t stream) {
  const float* states = (const float*)d_in[0];
  const float* actions= (const float*)d_in[1];
  const float* Ws     = (const float*)d_in[2];
  const float* bs     = (const float*)d_in[3];
  const float* Wsa    = (const float*)d_in[4];
  const float* bsa    = (const float*)d_in[5];
  const float* Wk     = (const float*)d_in[6];
  const float* Wsel   = (const float*)d_in[7];
  const float* Wv     = (const float*)d_in[8];
  const float* bv     = (const float*)d_in[9];
  const float* Wc1    = (const float*)d_in[10];
  const float* bc1    = (const float*)d_in[11];
  const float* Wc2    = (const float*)d_in[12];
  const float* bc2    = (const float*)d_in[13];
  const float* Wv1    = (const float*)d_in[14];
  const float* bv1    = (const float*)d_in[15];
  const float* Wv2    = (const float*)d_in[16];
  const float* bv2    = (const float*)d_in[17];
  float* outq = (float*)d_out;
  float* outv = outq + (size_t)N_AG * B_SZ;

  char* ws = (char*)d_ws;
  size_t off = 0;
  auto take = [&](size_t elems) -> _Float16* {
    _Float16* p = (_Float16*)(ws + off);
    off += ((elems * sizeof(_Float16) + 255) & ~(size_t)255);
    return p;
  };
  _Float16* states_h = take((size_t)N_AG * B_SZ * S_DIM);
  _Float16* sa_h     = take((size_t)N_AG * B_SZ * K_SA);
  _Float16* WsT_h    = take((size_t)N_AG * H_DIM * S_DIM);
  _Float16* WsaT_h   = take((size_t)N_AG * H_DIM * K_SA);
  _Float16* WkT_h    = take((size_t)H_DIM * H_DIM);
  _Float16* WselT_h  = take((size_t)H_DIM * H_DIM);
  _Float16* WvT_h    = take((size_t)H_DIM * H_DIM);
  _Float16* Wc1T_h   = take((size_t)N_AG * 2 * H_DIM * H_DIM);
  _Float16* Wv1T_h   = take((size_t)N_AG * 2 * H_DIM * H_DIM);
  _Float16* s_enc_h  = take((size_t)N_AG * B_SZ * H_DIM);
  _Float16* sa_enc_h = take((size_t)N_AG * B_SZ * H_DIM);
  _Float16* keys_h   = take((size_t)N_AG * B_SZ * H_DIM);
  _Float16* sels_h   = take((size_t)N_AG * B_SZ * H_DIM);
  _Float16* vals_h   = take((size_t)N_AG * B_SZ * H_DIM);
  _Float16* other_h  = take((size_t)N_AG * B_SZ * H_DIM);
  _Float16* hidc_h   = take((size_t)N_AG * B_SZ * H_DIM);
  _Float16* hidv_h   = take((size_t)N_AG * B_SZ * H_DIM);
  (void)ws_size; (void)in_sizes; (void)n_in; (void)out_size;

  // -- stage f16 copies (weights transposed for contiguous B fragments) --
  cvt_f16_kernel<<<4096, 256, 0, stream>>>(states, states_h, (size_t)N_AG * B_SZ * S_DIM);
  build_sa_kernel<<<4096, 256, 0, stream>>>(states, actions, sa_h);
  t_ws_kernel<<<1024, 256, 0, stream>>>(Ws,  WsT_h);
  t_wsa_kernel<<<1024, 256, 0, stream>>>(Wsa, WsaT_h);
  t_headw_kernel<<<1024, 256, 0, stream>>>(Wk,   WkT_h);
  t_headw_kernel<<<1024, 256, 0, stream>>>(Wsel, WselT_h);
  t_headw_kernel<<<1024, 256, 0, stream>>>(Wv,   WvT_h);
  t_w2h_kernel<<<4096, 256, 0, stream>>>(Wc1, Wc1T_h);
  t_w2h_kernel<<<4096, 256, 0, stream>>>(Wv1, Wv1T_h);

  const dim3 gblk(32, 8);
  const dim3 ggrd(256, N_AG);   // (8192/64)*(512/32)=2048 tiles / 8 waves

  const size_t sAB = (size_t)B_SZ * H_DIM;   // per-agent [B,512] stride

  // s_enc = lrelu(states @ Ws + bs)
  gemm_wmma_f16<<<ggrd, gblk, 0, stream>>>(states_h, (size_t)B_SZ * S_DIM, nullptr, 0,
      S_DIM, S_DIM, WsT_h, (size_t)H_DIM * S_DIM, bs, H_DIM, s_enc_h, sAB, B_SZ, H_DIM, 1);
  // sa_enc = lrelu(sa @ Wsa + bsa)
  gemm_wmma_f16<<<ggrd, gblk, 0, stream>>>(sa_h, (size_t)B_SZ * K_SA, nullptr, 0,
      K_SA, K_SA, WsaT_h, (size_t)H_DIM * K_SA, bsa, H_DIM, sa_enc_h, sAB, B_SZ, H_DIM, 1);
  // keys / sels / vals (shared weights across agents; output col = e*128+d)
  gemm_wmma_f16<<<ggrd, gblk, 0, stream>>>(sa_enc_h, sAB, nullptr, 0,
      H_DIM, H_DIM, WkT_h, 0, nullptr, 0, keys_h, sAB, B_SZ, H_DIM, 0);
  gemm_wmma_f16<<<ggrd, gblk, 0, stream>>>(sa_enc_h, sAB, nullptr, 0,
      H_DIM, H_DIM, WselT_h, 0, nullptr, 0, sels_h, sAB, B_SZ, H_DIM, 0);
  gemm_wmma_f16<<<ggrd, gblk, 0, stream>>>(sa_enc_h, sAB, nullptr, 0,
      H_DIM, H_DIM, WvT_h, 0, bv, 0, vals_h, sAB, B_SZ, H_DIM, 1);
  // attention -> other
  attention_kernel<<<(B_SZ * E_HD) / 8, gblk, 0, stream>>>(keys_h, sels_h, vals_h, other_h);
  // critic hidden: lrelu([sa_enc | other] @ Wc1 + bc1)
  gemm_wmma_f16<<<ggrd, gblk, 0, stream>>>(sa_enc_h, sAB, other_h, sAB,
      H_DIM, 2 * H_DIM, Wc1T_h, (size_t)H_DIM * 2 * H_DIM, bc1, H_DIM, hidc_h, sAB, B_SZ, H_DIM, 1);
  // value hidden: lrelu([s_enc | other] @ Wv1 + bv1)
  gemm_wmma_f16<<<ggrd, gblk, 0, stream>>>(s_enc_h, sAB, other_h, sAB,
      H_DIM, 2 * H_DIM, Wv1T_h, (size_t)H_DIM * 2 * H_DIM, bv1, H_DIM, hidv_h, sAB, B_SZ, H_DIM, 1);
  // q, v
  head_kernel<<<(N_AG * B_SZ) / 8, gblk, 0, stream>>>(hidc_h, hidv_h, Wc2, bc2, Wv2, bv2, outq, outv);
}
